// MixedLipMlp_43026982372013
// MI455X (gfx1250) — matrix operand order repaired and checked by reading, not compiled, not run on hardware
//
#include <hip/hip_runtime.h>
#include <hip/hip_bf16.h>

typedef __attribute__((ext_vector_type(16))) __bf16 v16bf;
typedef __attribute__((ext_vector_type(8)))  float  v8f;
typedef int v4i __attribute__((__vector_size__(16)));

#define B_ROWS     4096
#define LATENT_N   64
#define INPUT_N    256
#define HIDDEN_N   512
#define ACTIONS_N  12
#define EXPERTS_N  8
#define GATE_HN    128
#define IN_DIM_N   320   // LATENT + INPUT
#define INTER_N    576   // HIDDEN + LATENT

#if defined(__AMDGCN__) && __has_builtin(__builtin_amdgcn_global_load_async_to_lds_b128)
#define HAVE_ASYNC_LDS 1
#else
#define HAVE_ASYNC_LDS 0
#endif

__device__ __forceinline__ unsigned short f2bf(float f) {
  unsigned u = __float_as_uint(f);
  u = u + 0x7FFFu + ((u >> 16) & 1u);         // round-to-nearest-even
  return (unsigned short)(u >> 16);
}
__device__ __forceinline__ float bf2f(unsigned short h) {
  return __uint_as_float(((unsigned)h) << 16);
}
__device__ __forceinline__ float softplusf(float x) {
  return x > 20.f ? x : log1pf(expf(x));
}

// 16-byte global -> LDS copy; async pipe (ASYNCcnt) if available
__device__ __forceinline__ void copy16(const unsigned short* g, unsigned short* l) {
#if HAVE_ASYNC_LDS
  __builtin_amdgcn_global_load_async_to_lds_b128(
      (v4i*)const_cast<unsigned short*>(g), (v4i*)l, 0, 0);
#else
  *(uint4*)l = *(const uint4*)g;
#endif
}
__device__ __forceinline__ void async_wait0() {
#if HAVE_ASYNC_LDS
#if __has_builtin(__builtin_amdgcn_s_wait_asynccnt)
  __builtin_amdgcn_s_wait_asynccnt(0);
#else
  asm volatile("s_wait_asynccnt 0" ::: "memory");
#endif
#endif
}

// ---------------- weight prep: (E, IN, NOUT) f32 -> (E, NOUTP, IN) bf16, zero-padded ----------------
__global__ void transpose_w(const float* __restrict__ src, unsigned short* __restrict__ dst,
                            int IN, int NOUT, int NOUTP, int total) {
  int i = blockIdx.x * blockDim.x + threadIdx.x;
  if (i >= total) return;
  int per = NOUTP * IN;
  int e = i / per, rem = i % per;
  int n = rem / IN, k = rem % IN;
  float v = (n < NOUT) ? src[((size_t)e * IN + k) * NOUT + n] : 0.f;
  dst[i] = f2bf(v);
}

// ---------------- pack concat(z,c) into bf16 input buffers ----------------
__global__ void pack_inputs(const float* __restrict__ z, const float* __restrict__ c,
                            unsigned short* __restrict__ inp0,
                            unsigned short* __restrict__ inp1,
                            unsigned short* __restrict__ inp2) {
  int i = blockIdx.x * blockDim.x + threadIdx.x;
  if (i >= B_ROWS * IN_DIM_N) return;
  int b = i / IN_DIM_N, k = i % IN_DIM_N;
  float v = (k < LATENT_N) ? z[b * LATENT_N + k] : c[b * INPUT_N + (k - LATENT_N)];
  unsigned short h = f2bf(v);
  inp0[b * IN_DIM_N + k] = h;
  if (k < LATENT_N) {                 // z prefix of the 576-wide concat buffers
    inp1[b * INTER_N + k] = h;
    inp2[b * INTER_N + k] = h;
  }
}

// ---------------- Lipschitz row scales: min(softplus(c)/||W_row||_1, 1) ----------------
__global__ void gate_scales(const float* __restrict__ gw0, const float* __restrict__ gc0,
                            const float* __restrict__ gw1, const float* __restrict__ gc1,
                            const float* __restrict__ gw2, const float* __restrict__ gc2,
                            float* __restrict__ s0, float* __restrict__ s1,
                            float* __restrict__ s2) {
  int t = blockIdx.x * blockDim.x + threadIdx.x;
  if (t < GATE_HN) {
    const float* r = gw0 + (size_t)t * IN_DIM_N;
    float a = 0.f;
    for (int k = 0; k < IN_DIM_N; ++k) a += fabsf(r[k]);
    s0[t] = fminf(softplusf(gc0[0]) / a, 1.f);
  } else if (t < 2 * GATE_HN) {
    int n = t - GATE_HN;
    const float* r = gw1 + (size_t)n * GATE_HN;
    float a = 0.f;
    for (int k = 0; k < GATE_HN; ++k) a += fabsf(r[k]);
    s1[n] = fminf(softplusf(gc1[0]) / a, 1.f);
  } else if (t < 2 * GATE_HN + EXPERTS_N) {
    int n = t - 2 * GATE_HN;
    const float* r = gw2 + (size_t)n * GATE_HN;
    float a = 0.f;
    for (int k = 0; k < GATE_HN; ++k) a += fabsf(r[k]);
    s2[n] = fminf(softplusf(gc2[0]) / a, 1.f);
  }
}

// ---------------- gate layer 0: elu(x @ (s0*gw0)^T + gb0) ----------------
__global__ __launch_bounds__(GATE_HN)
void gate_l0(const float* __restrict__ z, const float* __restrict__ c,
             const float* __restrict__ gw0, const float* __restrict__ gb0,
             const float* __restrict__ s0, float* __restrict__ h0) {
  __shared__ float xs[IN_DIM_N];
  int b = blockIdx.x;
  for (int i = threadIdx.x; i < IN_DIM_N; i += GATE_HN)
    xs[i] = (i < LATENT_N) ? z[b * LATENT_N + i] : c[b * INPUT_N + i - LATENT_N];
  __syncthreads();
  int n = threadIdx.x;
  const float* wr = gw0 + (size_t)n * IN_DIM_N;
  float s = 0.f;
  for (int k = 0; k < IN_DIM_N; ++k) s = fmaf(xs[k], wr[k], s);
  s = s * s0[n] + gb0[n];
  h0[(size_t)b * GATE_HN + n] = s > 0.f ? s : (expf(s) - 1.f);
}

// ---------------- gate layer 1 ----------------
__global__ __launch_bounds__(GATE_HN)
void gate_l1(const float* __restrict__ h0,
             const float* __restrict__ gw1, const float* __restrict__ gb1,
             const float* __restrict__ s1, float* __restrict__ h1) {
  __shared__ float xs[GATE_HN];
  int b = blockIdx.x;
  xs[threadIdx.x] = h0[(size_t)b * GATE_HN + threadIdx.x];
  __syncthreads();
  int n = threadIdx.x;
  const float* wr = gw1 + (size_t)n * GATE_HN;
  float s = 0.f;
  for (int k = 0; k < GATE_HN; ++k) s = fmaf(xs[k], wr[k], s);
  s = s * s1[n] + gb1[n];
  h1[(size_t)b * GATE_HN + n] = s > 0.f ? s : (expf(s) - 1.f);
}

// ---------------- gate layer 2 + softmax -> coeff (B,8) ----------------
__global__ void gate_l2(const float* __restrict__ h1,
                        const float* __restrict__ gw2, const float* __restrict__ gb2,
                        const float* __restrict__ s2, float* __restrict__ coeff) {
  int b = blockIdx.x * blockDim.x + threadIdx.x;
  if (b >= B_ROWS) return;
  const float* x = h1 + (size_t)b * GATE_HN;
  float l[EXPERTS_N];
  float mx = -1e30f;
  for (int e = 0; e < EXPERTS_N; ++e) {
    const float* wr = gw2 + (size_t)e * GATE_HN;
    float s = 0.f;
    for (int k = 0; k < GATE_HN; ++k) s = fmaf(x[k], wr[k], s);
    s = s * s2[e] + gb2[e];
    l[e] = s;
    mx = fmaxf(mx, s);
  }
  float den = 0.f;
  for (int e = 0; e < EXPERTS_N; ++e) { l[e] = expf(l[e] - mx); den += l[e]; }
  float inv = 1.f / den;
  for (int e = 0; e < EXPERTS_N; ++e) coeff[(size_t)b * EXPERTS_N + e] = l[e] * inv;
}

// ---------------- MoE fused GEMM: out = sum_e coeff_e * (inp @ W_e) + coeff@bias ----------------
struct alignas(16) BF16x16 { unsigned short h[16]; };

// stage one (A 128x32, B BNx32) tile pair into LDS buffers (pure 16B copies)
template <int NT>
__device__ __forceinline__ void stage_tile(
    const unsigned short* __restrict__ inp, int IN,
    const unsigned short* __restrict__ Wt, int NOUTP,
    int mbase, int nbase, int e, int k0, int tid,
    unsigned short* Ab, unsigned short* Bb) {
  constexpr int LDA = 40, LDB = 40;
#pragma unroll
  for (int it = 0; it < 2; ++it) {          // 512 x 16B chunks for A
    int cidx = tid + it * 256;
    int m = cidx >> 2, kk = (cidx & 3) << 3;
    copy16(inp + (size_t)(mbase + m) * IN + k0 + kk, Ab + m * LDA + kk);
  }
  constexpr int BCH = NT * 16 * 4;          // BN rows x 4 chunks
  if (tid < BCH) {
    int n = tid >> 2, kk = (tid & 3) << 3;
    copy16(Wt + ((size_t)e * NOUTP + nbase + n) * IN + k0 + kk, Bb + n * LDB + kk);
  }
}

template <int NT>   // block = 8 waves x 16 rows = BM 128, BN = 16*NT
__global__ __launch_bounds__(256)
void moe_layer(const unsigned short* __restrict__ inp, int IN,
               const unsigned short* __restrict__ Wt,   // (E, NOUTP, IN) bf16, zero-padded
               int NOUTP,
               const float* __restrict__ bias, int NOUT,
               const float* __restrict__ coeff,
               unsigned short* __restrict__ nextInp,    // bf16 dest (B,576) at col 64, or null
               float* __restrict__ outF,                // f32 dest (B,NOUT), or null
               int applyElu) {
  constexpr int BM  = 128;
  constexpr int BN  = 16 * NT;
  constexpr int BK  = 32;
  constexpr int LDA = 40;   // padded ushort stride; 80B rows, 16B aligned
  constexpr int LDB = 40;
  __shared__ unsigned short Alds[2][BM * LDA];
  __shared__ unsigned short Blds[2][BN * LDB];   // Blds[n][k] (weights pre-transposed)

  const int tid   = threadIdx.x;
  const int wave  = tid >> 5;
  const int lane  = tid & 31;
  const int mbase = blockIdx.x * BM;
  const int nbase = blockIdx.y * BN;
  const int gmb   = mbase + wave * 16 + ((lane >> 4) << 3);  // first C row for this lane

  v8f acc[NT];
  v8f pacc[NT];
#pragma unroll
  for (int t = 0; t < NT; ++t) { acc[t] = {}; pacc[t] = {}; }

  const int ksteps = IN / BK;
  const int total  = EXPERTS_N * ksteps;

  // prologue: stage tile 0 into buffer 0
  stage_tile<NT>(inp, IN, Wt, NOUTP, mbase, nbase, 0, 0, tid,
                 &Alds[0][0], &Blds[0][0]);

  for (int s = 0; s < total; ++s) {
    const int buf = s & 1;
    async_wait0();        // this wave's async fills (tile s) have landed in LDS
    __syncthreads();      // everyone's fills landed; everyone done reading buf from s-2

    if (s + 1 < total) {  // overlap: async-fill the other buffer while we compute
      int e1 = (s + 1) / ksteps, k1 = ((s + 1) % ksteps) * BK;
      stage_tile<NT>(inp, IN, Wt, NOUTP, mbase, nbase, e1, k1, tid,
                     &Alds[buf ^ 1][0], &Blds[buf ^ 1][0]);
    }

    // A fragment: lanes 0-15 hold K{0-7,16-23}, lanes 16-31 K{8-15,24-31}
    BF16x16 ab;
    {
      const unsigned short* Ab = &Alds[buf][0];
      int row = wave * 16 + (lane & 15);
      int kh  = (lane >> 4) << 3;
      *(uint4*)&ab.h[0] = *(const uint4*)&Ab[row * LDA + kh];
      *(uint4*)&ab.h[8] = *(const uint4*)&Ab[row * LDA + kh + 16];
    }
    v16bf av = __builtin_bit_cast(v16bf, ab);
#pragma unroll
    for (int t = 0; t < NT; ++t) {
      BF16x16 bb;
      const unsigned short* Bb = &Blds[buf][0];
      int n  = t * 16 + (lane & 15);
      int kh = (lane >> 4) << 3;   // lanes 0-15: K{0-7,16-23}; lanes 16-31: K{8-15,24-31}
      *(uint4*)&bb.h[0] = *(const uint4*)&Bb[n * LDB + kh];
      *(uint4*)&bb.h[8] = *(const uint4*)&Bb[n * LDB + kh + 16];
      v16bf bv = __builtin_bit_cast(v16bf, bb);
      pacc[t] = __builtin_amdgcn_wmma_f32_16x16x32_bf16(
          false, av, false, bv, (short)0, pacc[t], false, false);
    }

    if (((s + 1) % ksteps) == 0) {          // expert finished: fold with its coefficient
      int e = s / ksteps;
      float cf[8];
#pragma unroll
      for (int r = 0; r < 8; ++r)
        cf[r] = coeff[(size_t)(gmb + r) * EXPERTS_N + e];
#pragma unroll
      for (int t = 0; t < NT; ++t) {
#pragma unroll
        for (int r = 0; r < 8; ++r)
          acc[t][r] = fmaf(cf[r], pacc[t][r], acc[t][r]);
        pacc[t] = {};
      }
    }
  }

  // ---- epilogue: + coeff@bias, ELU, write next concat buffer / final out ----
#pragma unroll
  for (int t = 0; t < NT; ++t) {
    int gn = nbase + t * 16 + (lane & 15);
    if (gn < NOUT) {
#pragma unroll
      for (int r = 0; r < 8; ++r) {
        int gm = gmb + r;
        float bm = 0.f;
#pragma unroll
        for (int e = 0; e < EXPERTS_N; ++e)
          bm = fmaf(coeff[(size_t)gm * EXPERTS_N + e], bias[(size_t)e * NOUT + gn], bm);
        float val = acc[t][r] + bm;
        if (applyElu) val = val > 0.f ? val : (expf(val) - 1.f);
        if (outF) outF[(size_t)gm * NOUT + gn] = val;
        else      nextInp[(size_t)gm * INTER_N + LATENT_N + gn] = f2bf(val);
      }
    }
  }
}

// =======================================================================
extern "C" void kernel_launch(void* const* d_in, const int* in_sizes, int n_in,
                              void* d_out, int out_size, void* d_ws, size_t ws_size,
                              hipStream_t stream) {
  const float* z   = (const float*)d_in[0];
  const float* c   = (const float*)d_in[1];
  const float* w0  = (const float*)d_in[2];
  const float* b0  = (const float*)d_in[3];
  const float* w1  = (const float*)d_in[4];
  const float* b1  = (const float*)d_in[5];
  const float* w2  = (const float*)d_in[6];
  const float* b2  = (const float*)d_in[7];
  const float* gw0 = (const float*)d_in[8];
  const float* gb0 = (const float*)d_in[9];
  const float* gc0 = (const float*)d_in[10];
  const float* gw1 = (const float*)d_in[11];
  const float* gb1 = (const float*)d_in[12];
  const float* gc1 = (const float*)d_in[13];
  const float* gw2 = (const float*)d_in[14];
  const float* gb2 = (const float*)d_in[15];
  const float* gc2 = (const float*)d_in[16];

  char* ws = (char*)d_ws;
  size_t off = 0;
  auto take = [&](size_t bytes) {
    char* p = ws + off;
    off += (bytes + 255) & ~(size_t)255;
    return p;
  };
  const int NT0 = EXPERTS_N * HIDDEN_N * IN_DIM_N;   // transposed w0: (8,512,320)
  const int NT1 = EXPERTS_N * HIDDEN_N * INTER_N;    // transposed w1: (8,512,576)
  const int NT2 = EXPERTS_N * 16 * INTER_N;          // transposed w2 padded: (8,16,576)
  unsigned short* w0t  = (unsigned short*)take((size_t)NT0 * 2);
  unsigned short* w1t  = (unsigned short*)take((size_t)NT1 * 2);
  unsigned short* w2t  = (unsigned short*)take((size_t)NT2 * 2);
  unsigned short* inp0 = (unsigned short*)take((size_t)B_ROWS * IN_DIM_N * 2);
  unsigned short* inp1 = (unsigned short*)take((size_t)B_ROWS * INTER_N * 2);
  unsigned short* inp2 = (unsigned short*)take((size_t)B_ROWS * INTER_N * 2);
  float* coeff = (float*)take((size_t)B_ROWS * EXPERTS_N * 4);
  float* h0    = (float*)take((size_t)B_ROWS * GATE_HN * 4);
  float* h1    = (float*)take((size_t)B_ROWS * GATE_HN * 4);
  float* s0    = (float*)take(GATE_HN * 4);
  float* s1    = (float*)take(GATE_HN * 4);
  float* s2    = (float*)take(EXPERTS_N * 4);

  // 1) weights -> bf16, transposed to (E, NOUTP, IN), layer2 zero-padded to 16 cols
  transpose_w<<<(NT0 + 255) / 256, 256, 0, stream>>>(w0, w0t, IN_DIM_N, HIDDEN_N, HIDDEN_N, NT0);
  transpose_w<<<(NT1 + 255) / 256, 256, 0, stream>>>(w1, w1t, INTER_N, HIDDEN_N, HIDDEN_N, NT1);
  transpose_w<<<(NT2 + 255) / 256, 256, 0, stream>>>(w2, w2t, INTER_N, ACTIONS_N, 16, NT2);

  // 2) pack concat(z,c) + z prefixes
  pack_inputs<<<(B_ROWS * IN_DIM_N + 255) / 256, 256, 0, stream>>>(z, c, inp0, inp1, inp2);

  // 3) gate network -> coeff
  gate_scales<<<1, 288, 0, stream>>>(gw0, gc0, gw1, gc1, gw2, gc2, s0, s1, s2);
  gate_l0<<<B_ROWS, GATE_HN, 0, stream>>>(z, c, gw0, gb0, s0, h0);
  gate_l1<<<B_ROWS, GATE_HN, 0, stream>>>(h0, gw1, gb1, s1, h1);
  gate_l2<<<B_ROWS / 256, 256, 0, stream>>>(h1, gw2, gb2, s2, coeff);

  // 4) MoE layers (WMMA bf16, async double-buffered LDS staging)
  moe_layer<4><<<dim3(B_ROWS / 128, HIDDEN_N / 64), 256, 0, stream>>>(
      inp0, IN_DIM_N, w0t, HIDDEN_N, b0, HIDDEN_N, coeff, inp1, nullptr, 1);
  moe_layer<4><<<dim3(B_ROWS / 128, HIDDEN_N / 64), 256, 0, stream>>>(
      inp1, INTER_N, w1t, HIDDEN_N, b1, HIDDEN_N, coeff, inp2, nullptr, 1);
  moe_layer<1><<<dim3(B_ROWS / 128, 1), 256, 0, stream>>>(
      inp2, INTER_N, w2t, 16, b2, ACTIONS_N, coeff, nullptr, (float*)d_out, 0);
}
